// GNNTransformer_300647711093
// MI455X (gfx1250) — compile-verified, matrix operand order checked by reading, not run on hardware
//
#include <hip/hip_runtime.h>
#include <hip/hip_bf16.h>

typedef __attribute__((ext_vector_type(16))) _Float16 v16h;
typedef __attribute__((ext_vector_type(8)))  float    v8f;
typedef __attribute__((ext_vector_type(4)))  int      v4i;

#define HD 128
#define IND 64
#define MT 64    // M-rows per k_node block
#define RS 136   // LDS row stride (halves) for 128-wide tiles
#define RS64 72  // LDS row stride (halves) for 64-wide tiles

#if defined(__has_builtin)
#if __has_builtin(__builtin_amdgcn_global_load_async_to_lds_b128) && \
    __has_builtin(__builtin_amdgcn_s_wait_asynccnt)
#define USE_ASYNC_LDS 1
#endif
#endif
#ifndef USE_ASYNC_LDS
#define USE_ASYNC_LDS 0
#endif

static __device__ __forceinline__ v8f wmma16(v16h a, v16h b, v8f c) {
  return __builtin_amdgcn_wmma_f32_16x16x32_f16(false, a, false, b, (short)0, c, false, false);
}

// Wave32 WMMA 16x16x32 f16 fragment half-row from a row-major f16 buffer.
// Caller passes p = base + row*stride + kb + off, off = (lane&16)?8:0.
// Lane holds halves {p[0..7], p[16..23]}  (ISA 7.12.2 16-bit A layout; B mirrors, n as "row").
static __device__ __forceinline__ v16h load_frag(const _Float16* p) {
  v16h f;
#pragma unroll
  for (int i = 0; i < 8; ++i) { f[i] = p[i]; f[8 + i] = p[16 + i]; }
  return f;
}

static __device__ __forceinline__ float elu_f(float x) {
  float e = __expf(x) - 1.f;       // branchless: both sides cheap, select via cndmask
  return x > 0.f ? x : e;
}

// ---------------------------------------------------------------- utility ---
__global__ void k_zero(float* __restrict__ p, long n) {
  long i = (long)blockIdx.x * blockDim.x + threadIdx.x;
  if (i < n) p[i] = 0.f;
}

// in: [B][K][N] f32  ->  out: [B][N][K] f16 (transposed per batch slice)
__global__ void k_cvt(const float* __restrict__ in, _Float16* __restrict__ out,
                      int B, int K, int N) {
  long total = (long)B * K * N;
  long i = (long)blockIdx.x * blockDim.x + threadIdx.x;
  if (i >= total) return;
  int n = (int)(i % N);
  int k = (int)((i / N) % K);
  int b = (int)(i / ((long)N * K));
  out[((long)b * N + n) * K + k] = (_Float16)in[i];
}

// ------------------------------------------------- input projection (WMMA) ---
// Hout[g][n] = elu( sum_k X[g][k]*W[k][n] + bias[n] ),  K=64, N=128
__global__ __launch_bounds__(256) void k_inproj(const float* __restrict__ X,
                                                const _Float16* __restrict__ WT, // [128][64]
                                                const float* __restrict__ bias,
                                                float* __restrict__ Hout, int M) {
  __shared__ _Float16 sA[16 * RS64];
  int m0 = blockIdx.x * 16;
  int t = threadIdx.x;
  for (int i = t; i < 16 * IND; i += 256) {
    int r = i >> 6, c = i & 63;
    int g = m0 + r;
    sA[r * RS64 + c] = (_Float16)((g < M) ? X[(long)g * IND + c] : 0.f);
  }
  __syncthreads();
  int lane = t & 31;
  int n0 = (t >> 5) * 16;
  int col = lane & 15;
  int off = (lane & 16) ? 8 : 0;
  v8f acc = {};
#pragma unroll
  for (int kb = 0; kb < IND; kb += 32) {
    v16h a = load_frag(sA + col * RS64 + kb + off);
    v16h b = load_frag(WT + (long)(n0 + col) * IND + kb + off);
    acc = wmma16(a, b, acc);
  }
  int n = n0 + col;
  float bv = bias[n];
#pragma unroll
  for (int r = 0; r < 8; ++r) {
    int g = m0 + r + off;
    if (g < M) Hout[(long)g * HD + n] = elu_f(acc[r] + bv);
  }
}

// ------------------------------------------------------------ edge message ---
// One wave per edge; lane handles 4 feature dims (float4).
// msg = relu(h_src + ea@eW + eb); agg[dst] += msg (native f32 atomics, L2-resident)
__global__ __launch_bounds__(256) void k_edge(const int* __restrict__ src,
                                              const int* __restrict__ dst,
                                              const float* __restrict__ ea,    // [E][3]
                                              const float* __restrict__ eW,    // [3][128]
                                              const float* __restrict__ eb,    // [128]
                                              const float* __restrict__ hs,
                                              float* __restrict__ agg, int E) {
  int w = (blockIdx.x * blockDim.x + threadIdx.x) >> 5;
  int lane = threadIdx.x & 31;
  if (w >= E) return;
  int s = src[w], d = dst[w];
  float a0 = ea[(long)w * 3 + 0];
  float a1 = ea[(long)w * 3 + 1];
  float a2 = ea[(long)w * 3 + 2];
  int c = lane * 4;
  float4 w0 = *(const float4*)(eW + 0 * HD + c);
  float4 w1 = *(const float4*)(eW + 1 * HD + c);
  float4 w2 = *(const float4*)(eW + 2 * HD + c);
  float4 bb = *(const float4*)(eb + c);
  float4 hv = *(const float4*)(hs + (long)s * HD + c);
  float m0 = hv.x + a0 * w0.x + a1 * w1.x + a2 * w2.x + bb.x;
  float m1 = hv.y + a0 * w0.y + a1 * w1.y + a2 * w2.y + bb.y;
  float m2 = hv.z + a0 * w0.z + a1 * w1.z + a2 * w2.z + bb.z;
  float m3 = hv.w + a0 * w0.w + a1 * w1.w + a2 * w2.w + bb.w;
  float* p = agg + (long)d * HD + c;
  if (m0 > 0.f) unsafeAtomicAdd(p + 0, m0);
  if (m1 > 0.f) unsafeAtomicAdd(p + 1, m1);
  if (m2 > 0.f) unsafeAtomicAdd(p + 2, m2);
  if (m3 > 0.f) unsafeAtomicAdd(p + 3, m3);
}

// ------------------------------------------- fused node MLP (double WMMA) ---
// z = h + agg ; y = relu((z@W1 + b1)*bn_inv*g1 + be1) ; out (=|+=) y@W2 + b2
// 64-row M-tile per block, 8 waves; B-fragments reused across 4 M-subtiles.
__global__ __launch_bounds__(256) void k_node(const float* __restrict__ ht,
                                              const float* __restrict__ agg,
                                              const _Float16* __restrict__ W1t, // [128][128]
                                              const float* __restrict__ b1,
                                              const float* __restrict__ g1,
                                              const float* __restrict__ be1,
                                              const _Float16* __restrict__ W2t, // [128][128]
                                              const float* __restrict__ b2,
                                              float* __restrict__ outacc, int M,
                                              int accum) {
  __shared__ _Float16 sZ[MT * RS];
  __shared__ union {
    float stage[MT * HD];      // async-staged h tile (dead after sZ is built)
    _Float16 y[MT * RS];       // relu'd GEMM1 output tile (f16)
  } uS;
  int m0 = blockIdx.x * MT;
  int t = threadIdx.x;
  int lane = t & 31;
  int n0 = (t >> 5) * 16;
  int col = lane & 15;
  int off = (lane & 16) ? 8 : 0;
  int n = n0 + col;

  // warm L2/L0 for this wave's weight slices while we stage the z tile
  __builtin_prefetch(W1t + (long)n * HD, 0, 3);
  __builtin_prefetch(W2t + (long)n * HD, 0, 3);

#if USE_ASYNC_LDS
  // Stage h[t] tile into LDS with the async DMA path (ASYNCcnt), agg read direct.
  for (int i = t; i < MT * HD / 4; i += 256) {
    int r = i >> 5;                 // 32 float4 chunks per 128-wide row
    int c4 = (i & 31) * 4;
    int g = m0 + r;
    int gc = (g < M) ? g : (M - 1); // clamp; padded rows zeroed below
    __builtin_amdgcn_global_load_async_to_lds_b128(
        (__attribute__((address_space(1))) v4i*)(ht + (long)gc * HD + c4),
        (__attribute__((address_space(3))) v4i*)(uS.stage + r * HD + c4), 0, 0);
  }
  __builtin_amdgcn_s_wait_asynccnt(0);
  __syncthreads();
  for (int i = t; i < MT * HD; i += 256) {
    int r = i >> 7, c = i & 127;
    int g = m0 + r;
    float v = (g < M) ? (uS.stage[i] + agg[(long)g * HD + c]) : 0.f;
    sZ[r * RS + c] = (_Float16)v;
  }
#else
  for (int i = t; i < MT * HD; i += 256) {
    int r = i >> 7, c = i & 127;
    int g = m0 + r;
    float v = (g < M) ? (ht[(long)g * HD + c] + agg[(long)g * HD + c]) : 0.f;
    sZ[r * RS + c] = (_Float16)v;
  }
#endif
  __syncthreads();

  // GEMM1: z @ W1  (K=128); B fragment reused across 4 M-subtiles
  v8f c1[4] = {{}, {}, {}, {}};
#pragma unroll
  for (int kb = 0; kb < HD; kb += 32) {
    v16h b = load_frag(W1t + (long)n * HD + kb + off);
#pragma unroll
    for (int mt = 0; mt < 4; ++mt) {
      v16h a = load_frag(sZ + (mt * 16 + col) * RS + kb + off);
      c1[mt] = wmma16(a, b, c1[mt]);
    }
  }

  // BN (eval) + ReLU epilogue, stage y tiles as f16 in LDS (reuses stage memory)
  const float bninv = 0.9999950000374997f; // 1/sqrt(1+1e-5)
  float bb1 = b1[n], gg1 = g1[n], bbe = be1[n];
#pragma unroll
  for (int mt = 0; mt < 4; ++mt) {
#pragma unroll
    for (int r = 0; r < 8; ++r) {
      float y = (c1[mt][r] + bb1) * bninv * gg1 + bbe;
      uS.y[(mt * 16 + r + off) * RS + n] = (_Float16)fmaxf(y, 0.f);
    }
  }
  __syncthreads();

  // GEMM2: y @ W2  (K=128)
  v8f c2[4] = {{}, {}, {}, {}};
#pragma unroll
  for (int kb = 0; kb < HD; kb += 32) {
    v16h b = load_frag(W2t + (long)n * HD + kb + off);
#pragma unroll
    for (int mt = 0; mt < 4; ++mt) {
      v16h a = load_frag(uS.y + (mt * 16 + col) * RS + kb + off);
      c2[mt] = wmma16(a, b, c2[mt]);
    }
  }
  float bb2 = b2[n];
#pragma unroll
  for (int mt = 0; mt < 4; ++mt) {
#pragma unroll
    for (int r = 0; r < 8; ++r) {
      int g = m0 + mt * 16 + r + off;
      if (g < M) {
        float v = c2[mt][r] + bb2;
        if (accum) outacc[(long)g * HD + n] += v;
        else       outacc[(long)g * HD + n] = v;
      }
    }
  }
}

// ----------------------------------------------------------- residual+ELU ---
__global__ void k_resid(const float* __restrict__ outacc, float* __restrict__ h, long n) {
  long i = (long)blockIdx.x * blockDim.x + threadIdx.x;
  if (i < n) h[i] = elu_f(outacc[i] + h[i]);
}

// ===========================================================================
extern "C" void kernel_launch(void* const* d_in, const int* in_sizes, int n_in,
                              void* d_out, int out_size, void* d_ws, size_t ws_size,
                              hipStream_t stream) {
  (void)in_sizes; (void)n_in; (void)out_size; (void)ws_size;
  const int L = 3, R = 7;
  const int NT[3] = {10000, 40000, 100000};
  const int E[7]  = {50000, 100000, 100000, 150000, 200000, 200000, 200000};
  const int ST[7] = {0, 0, 1, 1, 1, 2, 2};
  const int DT[7] = {0, 1, 0, 1, 2, 1, 2};

  const float* x[3]  = {(const float*)d_in[0], (const float*)d_in[1], (const float*)d_in[2]};
  const int*   srcs[7]; const int* dsts[7]; const float* eas[7];
  for (int r = 0; r < R; ++r) {
    srcs[r] = (const int*)d_in[3 + 3 * r];
    dsts[r] = (const int*)d_in[4 + 3 * r];
    eas[r]  = (const float*)d_in[5 + 3 * r];
  }
  const float* linW[3] = {(const float*)d_in[24], (const float*)d_in[26], (const float*)d_in[28]};
  const float* linb[3] = {(const float*)d_in[25], (const float*)d_in[27], (const float*)d_in[29]};
  const float* eW  = (const float*)d_in[30];
  const float* eb  = (const float*)d_in[31];
  const float* W1  = (const float*)d_in[32];
  const float* b1  = (const float*)d_in[33];
  const float* g1  = (const float*)d_in[34];
  const float* be1 = (const float*)d_in[35];
  const float* W2  = (const float*)d_in[36];
  const float* b2  = (const float*)d_in[37];

  // hidden states live in d_out (it is exactly the concatenated outputs)
  float* h[3];
  h[0] = (float*)d_out;
  h[1] = h[0] + (long)NT[0] * HD;
  h[2] = h[1] + (long)NT[1] * HD;
  const long totF = ((long)NT[0] + NT[1] + NT[2]) * HD; // 19.2M floats

  // workspace carve-up
  char* w = (char*)d_ws;
  float* outAll = (float*)w;                 w += totF * sizeof(float);             // 76.8 MB
  float* outp[3] = {outAll, outAll + (long)NT[0] * HD, outAll + ((long)NT[0] + NT[1]) * HD};
  float* aggs = (float*)w;                   w += (long)NT[2] * HD * sizeof(float); // 51.2 MB
  _Float16* linWT = (_Float16*)w;            w += 3L * HD * IND * sizeof(_Float16);
  _Float16* W1t = (_Float16*)w;              w += (long)L * R * HD * HD * sizeof(_Float16);
  _Float16* W2t = (_Float16*)w;              w += (long)L * R * HD * HD * sizeof(_Float16);

  // 1) convert weights to transposed f16
  for (int t = 0; t < 3; ++t) {
    int tot = IND * HD;
    k_cvt<<<(tot + 255) / 256, 256, 0, stream>>>(linW[t], linWT + (long)t * HD * IND, 1, IND, HD);
  }
  {
    long tot = (long)L * R * HD * HD;
    k_cvt<<<(int)((tot + 255) / 256), 256, 0, stream>>>(W1, W1t, L * R, HD, HD);
    k_cvt<<<(int)((tot + 255) / 256), 256, 0, stream>>>(W2, W2t, L * R, HD, HD);
  }

  // 2) input projections  h[t] = elu(x @ linW + linb)
  for (int t = 0; t < 3; ++t)
    k_inproj<<<(NT[t] + 15) / 16, 256, 0, stream>>>(x[t], linWT + (long)t * HD * IND,
                                                    linb[t], h[t], NT[t]);

  // 3) message-passing layers
  for (int l = 0; l < L; ++l) {
    for (int r = 0; r < R; ++r) {
      int s = ST[r], t = DT[r];
      long an = (long)NT[t] * HD;
      k_zero<<<(int)((an + 255) / 256), 256, 0, stream>>>(aggs, an);
      k_edge<<<(E[r] + 7) / 8, 256, 0, stream>>>(srcs[r], dsts[r], eas[r],
                                                 eW + (long)(l * R + r) * 3 * HD,
                                                 eb + (long)(l * R + r) * HD,
                                                 h[s], aggs, E[r]);
      long woff = (long)(l * R + r) * HD * HD;
      long voff = (long)(l * R + r) * HD;
      // first relation targeting each type this layer overwrites out[t]
      int accum = !(r == 0 || r == 1 || r == 4);
      k_node<<<(NT[t] + MT - 1) / MT, 256, 0, stream>>>(h[t], aggs,
                                                        W1t + woff, b1 + voff, g1 + voff, be1 + voff,
                                                        W2t + woff, b2 + voff,
                                                        outp[t], NT[t], accum);
    }
    k_resid<<<(int)((totF + 255) / 256), 256, 0, stream>>>(outAll, h[0], totF);
  }
  // h (== d_out) now holds (h_core, h_agg, h_access)
}
